// HandwritingModel_22230750724448
// MI455X (gfx1250) — matrix operand order repaired and checked by reading, not compiled, not run on hardware
//
#include <hip/hip_runtime.h>

// ---------------------------------------------------------------------------
// Handwriting model (3x LSTM-512 + attention window + MDN) for gfx1250.
// Recurrent GEMMs run on v_wmma_f32_16x16x32_bf16 (bf16 in, f32 accum).
// Input projections are folded into the recurrent GEMM along K.
// ---------------------------------------------------------------------------

typedef unsigned short u16t;
typedef __attribute__((ext_vector_type(16))) __bf16        v16bf;
typedef __attribute__((ext_vector_type(8)))  float         v8f;
typedef __attribute__((ext_vector_type(4)))  unsigned int  uint4v;

constexpr int kB   = 64;    // batch
constexpr int kT   = 800;   // timesteps
constexpr int kH   = 512;   // hidden
constexpr int kAug = 608;   // 512 (h0) + 80 (window) + 3 (stroke) + 13 pad
constexpr int KH_TILES = kH / 32;   // 16 K-tiles for the h-part

union FragU { uint4v q[2]; v16bf v; };

__device__ __forceinline__ u16t f2bf(float f) {
  unsigned u = __float_as_uint(f);
  return (u16t)((u + 0x7FFFu + ((u >> 16) & 1u)) >> 16);   // RNE
}
__device__ __forceinline__ float bf2f(u16t h) {
  return __uint_as_float(((unsigned)h) << 16);
}
__device__ __forceinline__ float sigm(float x) { return 1.0f / (1.0f + __expf(-x)); }
__device__ __forceinline__ float tanhfast(float x) {
  float ax = fabsf(x);
  if (ax > 8.0f) return copysignf(1.0f, x);
  float e = __expf(2.0f * x);
  return (e - 1.0f) / (e + 1.0f);
}

// A fragment (16x32 bf16, M-rows = batch) per ISA 16-bit A layout:
// lanes 0-15: row m0+l, K = k0+0..7 and k0+16..23 ; lanes 16-31: K = +8.
__device__ __forceinline__ v16bf loadA(const u16t* base, int stride, int m0,
                                       int kc, int lane) {
  int ls = lane & 15, lh = lane >> 4;
  const u16t* p = base + (size_t)(m0 + ls) * stride + kc + lh * 8;
  FragU f;
  f.q[0] = *(const uint4v*)(p);
  f.q[1] = *(const uint4v*)(p + 16);
  return f.v;
}
// B fragment: pre-swizzled weights, lane-contiguous (32 bytes per lane).
__device__ __forceinline__ v16bf loadB(const u16t* pack, int frag, int lane) {
  const u16t* p = pack + (size_t)frag * 512 + lane * 16;
  FragU f;
  f.q[0] = *(const uint4v*)(p);
  f.q[1] = *(const uint4v*)(p + 8);
  return f.v;
}

// --------------------------- utility kernels --------------------------------

__global__ void zero_kernel(unsigned* __restrict__ p, int nwords) {
  int i = blockIdx.x * blockDim.x + threadIdx.x;
  if (i < nwords) p[i] = 0u;
}

// Pack [Ktot x 2048] weight (rows 0..Ku-1 from U, Ku..Ku+Kw-1 from W, rest 0)
// into WMMA B-fragment layout. One wave per 32(K) x 16(N) fragment.
__global__ void pack_weights_kernel(const float* __restrict__ U, int Ku,
                                    const float* __restrict__ W, int Kw,
                                    u16t* __restrict__ dst) {
  int f = blockIdx.x;
  int lane = threadIdx.x;
  int kb = f >> 7, ntile = f & 127;
  int k0 = kb * 32;
  int n = ntile * 16 + (lane & 15);
  int lh = lane >> 4;
  u16t* o = dst + (size_t)f * 512 + lane * 16;
#pragma unroll
  for (int j = 0; j < 16; ++j) {
    int k = (j < 8) ? (k0 + lh * 8 + j) : (k0 + 16 + lh * 8 + (j - 8));
    float v = 0.0f;
    if (k < Ku) v = U[(size_t)k * 2048 + n];
    else if (k - Ku < Kw) v = W[(size_t)(k - Ku) * 2048 + n];
    o[j] = f2bf(v);
  }
}

// aug[t][b][592..594] = stroke, cols 595..607 = 0
__global__ void init_aug_kernel(const float* __restrict__ stroke,
                                u16t* __restrict__ aug) {
  int tid = blockIdx.x * blockDim.x + threadIdx.x;
  if (tid >= kB * kT) return;
  int t = tid % kT, b = tid / kT;
  u16t* row = aug + ((size_t)t * kB + b) * kAug;
  const float* s = stroke + ((size_t)b * kT + t) * 3;
  row[592] = f2bf(s[0]);
  row[593] = f2bf(s[1]);
  row[594] = f2bf(s[2]);
  for (int c = 595; c < kAug; ++c) row[c] = 0;
}

// ----------------------------- LSTM step ------------------------------------
// grid 128 blocks (4 m-tiles x 32 n-tiles), 128 threads = 4 waves.
// Wave g computes gate g's 16x16 tile of z = [h ; x_t] @ [U ; W] via WMMA,
// gates combined through LDS, fused cell update + bf16 h re-emit.
__global__ __launch_bounds__(128) void lstm_step_kernel(
    const u16t* __restrict__ packW, int kTilesX,
    const u16t* __restrict__ hread, u16t* __restrict__ hwrite,
    float* __restrict__ cbuf,
    const u16t* __restrict__ xrow, int xstride,
    u16t* __restrict__ dstrow, int dstride,
    const float* __restrict__ bias,
    const float* __restrict__ W0x, const float* __restrict__ stroke, int t) {
  __shared__ float zsm[4][256];
  const int tid = threadIdx.x;
  const int lane = tid & 31;
  const int g = tid >> 5;                 // gate = wave id
  const int m0 = (blockIdx.x >> 5) * 16;  // batch tile
  const int n0 = (blockIdx.x & 31) * 16;  // hidden tile
  const int nTileG = g * 32 + (blockIdx.x & 31);

  v8f acc = {0.f, 0.f, 0.f, 0.f, 0.f, 0.f, 0.f, 0.f};

#pragma unroll
  for (int kb = 0; kb < KH_TILES; ++kb) {
    v16bf a = loadA(hread, kH, m0, kb * 32, lane);
    v16bf b = loadB(packW, kb * 128 + nTileG, lane);
    acc = __builtin_amdgcn_wmma_f32_16x16x32_bf16(false, a, false, b,
                                                  (short)0, acc, false, false);
  }
  for (int kb = 0; kb < kTilesX; ++kb) {
    if (kb + 1 < kTilesX)
      __builtin_prefetch(packW + ((size_t)(KH_TILES + kb + 1) * 128 + nTileG) * 512 +
                             lane * 16, 0, 1);
    v16bf a = loadA(xrow, xstride, m0, kb * 32, lane);
    v16bf b = loadB(packW, (KH_TILES + kb) * 128 + nTileG, lane);
    acc = __builtin_amdgcn_wmma_f32_16x16x32_bf16(false, a, false, b,
                                                  (short)0, acc, false, false);
  }

  {  // C/D layout: lane l, VGPR r -> m = r + (l>=16)*8, n = l&15
    int ls = lane & 15, lh = lane >> 4;
#pragma unroll
    for (int r = 0; r < 8; ++r) zsm[g][(r + lh * 8) * 16 + ls] = acc[r];
  }
  __syncthreads();

#pragma unroll
  for (int rep = 0; rep < 2; ++rep) {
    int e = tid + rep * 128;
    int bidx = m0 + (e >> 4);
    int n = n0 + (e & 15);
    float zi = zsm[0][e] + bias[n];
    float zf = zsm[1][e] + bias[512 + n];
    float zg = zsm[2][e] + bias[1024 + n];
    float zo = zsm[3][e] + bias[1536 + n];
    if (W0x) {  // LSTM0: fold x(3) @ W0 scalar-wise
      const float* xr = stroke + ((size_t)bidx * kT + t) * 3;
      float x0 = xr[0], x1 = xr[1], x2 = xr[2];
      zi += x0 * W0x[n] + x1 * W0x[2048 + n] + x2 * W0x[4096 + n];
      zf += x0 * W0x[512 + n] + x1 * W0x[2048 + 512 + n] + x2 * W0x[4096 + 512 + n];
      zg += x0 * W0x[1024 + n] + x1 * W0x[2048 + 1024 + n] + x2 * W0x[4096 + 1024 + n];
      zo += x0 * W0x[1536 + n] + x1 * W0x[2048 + 1536 + n] + x2 * W0x[4096 + 1536 + n];
    }
    size_t ci = (size_t)bidx * kH + n;
    float cold = cbuf[ci];
    float cn = sigm(zf) * cold + sigm(zi) * tanhfast(zg);
    float h = sigm(zo) * tanhfast(cn);
    cbuf[ci] = cn;
    u16t hb = f2bf(h);
    hwrite[ci] = hb;
    dstrow[(size_t)bidx * dstride + n] = hb;
  }
}

// ----------------------- attention pipeline ---------------------------------

// abk = exp(lstm0_out @ Wa + ba); alpha/beta stored, koff stored into kap.
__global__ void abk_kernel(const u16t* __restrict__ aug,
                           const float* __restrict__ Wa,
                           const float* __restrict__ ba,
                           float* __restrict__ alpha, float* __restrict__ beta,
                           float* __restrict__ kap) {
  int tid = blockIdx.x * blockDim.x + threadIdx.x;
  if (tid >= kB * kT * 30) return;
  int j = tid % 30;
  int bt = tid / 30;
  int t = bt % kT, b = bt / kT;
  const u16t* row = aug + ((size_t)t * kB + b) * kAug;
  float z = ba[j];
  for (int c = 0; c < kH; ++c) z += bf2f(row[c]) * Wa[c * 30 + j];
  float e = __expf(z);
  int k = j % 10, which = j / 10;
  size_t idx = ((size_t)b * 10 + k) * kT + t;
  if (which == 0) alpha[idx] = e;
  else if (which == 1) beta[idx] = e;
  else kap[idx] = e;
}

__global__ void cumsum_kernel(const float* __restrict__ kappa0,
                              float* __restrict__ kap) {
  int tid = blockIdx.x * blockDim.x + threadIdx.x;
  if (tid >= kB * 10) return;
  int b = tid / 10, k = tid % 10;
  float run = kappa0[b * 10 + k];
  float* p = kap + ((size_t)b * 10 + k) * kT;
  for (int t = 0; t < kT; ++t) { run += p[t]; p[t] = run; }
}

// phi + window fused; writes aug cols 512..591 (bf16).
__global__ void window_kernel(const float* __restrict__ alpha,
                              const float* __restrict__ beta,
                              const float* __restrict__ kap,
                              const float* __restrict__ chars,
                              u16t* __restrict__ aug) {
  __shared__ float phi[80];
  int bt = blockIdx.x;
  int t = bt % kT, b = bt / kT;
  int u = threadIdx.x;  // 80 threads
  float uf = (float)u, p = 0.0f;
#pragma unroll
  for (int k = 0; k < 10; ++k) {
    size_t idx = ((size_t)b * 10 + k) * kT + t;
    float d = kap[idx] - uf;
    p += alpha[idx] * __expf(-beta[idx] * d * d);
  }
  phi[u] = p;
  __syncthreads();
  int c = threadIdx.x;
  float wv = 0.0f;
  for (int uu = 0; uu < 80; ++uu) wv += phi[uu] * chars[((size_t)b * 80 + uu) * 80 + c];
  aug[((size_t)t * kB + b) * kAug + 512 + c] = f2bf(wv);
}

// ------------------------------- MDN head -----------------------------------

__global__ void mdn_kernel(const u16t* __restrict__ l2o,
                           const float* __restrict__ Wm,
                           const float* __restrict__ bm,
                           float* __restrict__ out) {
  __shared__ float z[121];
  int bt = blockIdx.x;
  int t = bt % kT, b = bt / kT;
  int j = threadIdx.x;  // 128
  if (j < 121) {
    const u16t* row = l2o + ((size_t)t * kB + b) * kH;
    float s = bm[j];
    for (int c = 0; c < kH; ++c) s += bf2f(row[c]) * Wm[c * 121 + j];
    z[j] = s;
  }
  __syncthreads();
  if (j >= 121) return;
  float v;
  if (j == 0) {
    v = 1.0f / (1.0f + __expf(z[0]));          // sigmoid(-eos)
  } else if (j < 21) {                         // softmax over pi_hat
    float m = z[1];
    for (int i = 2; i < 21; ++i) m = fmaxf(m, z[i]);
    float s = 0.0f;
    for (int i = 1; i < 21; ++i) s += __expf(z[i] - m);
    v = __expf(z[j] - m) / s;
  } else if (j < 61) {
    v = z[j];                                  // mu1, mu2
  } else if (j < 101) {
    v = __expf(z[j]);                          // exp(s1), exp(s2)
  } else {
    v = tanhfast(z[j]);                        // tanh(rho)
  }
  out[((size_t)b * kT + t) * 121 + j] = v;
}

// ------------------------------ launcher ------------------------------------

extern "C" void kernel_launch(void* const* d_in, const int* in_sizes, int n_in,
                              void* d_out, int out_size, void* d_ws, size_t ws_size,
                              hipStream_t stream) {
  (void)in_sizes; (void)n_in; (void)out_size; (void)ws_size;
  const float* stroke = (const float*)d_in[0];
  const float* chars  = (const float*)d_in[1];
  const float* kappa0 = (const float*)d_in[2];
  const float* W0 = (const float*)d_in[3];
  const float* U0 = (const float*)d_in[4];
  const float* b0 = (const float*)d_in[5];
  const float* W1 = (const float*)d_in[6];
  const float* U1 = (const float*)d_in[7];
  const float* b1 = (const float*)d_in[8];
  const float* W2 = (const float*)d_in[9];
  const float* U2 = (const float*)d_in[10];
  const float* b2 = (const float*)d_in[11];
  const float* Wa = (const float*)d_in[12];
  const float* ba = (const float*)d_in[13];
  const float* Wm = (const float*)d_in[14];
  const float* bm = (const float*)d_in[15];
  float* out = (float*)d_out;

  char* w = (char*)d_ws;
  auto take = [&](size_t bytes) -> char* {
    char* p = w;
    w += (bytes + 1023) & ~(size_t)1023;
    return p;
  };
  u16t* aug = (u16t*)take((size_t)kT * kB * kAug * 2);   // [T,B,608] bf16
  u16t* l1o = (u16t*)take((size_t)kT * kB * kH * 2);     // [T,B,512] bf16
  u16t* l2o = (u16t*)take((size_t)kT * kB * kH * 2);     // [T,B,512] bf16
  u16t* pk0 = (u16t*)take((size_t)(512 / 32) * 128 * 1024);
  u16t* pk1 = (u16t*)take((size_t)(1120 / 32) * 128 * 1024);
  u16t* pk2 = (u16t*)take((size_t)(1024 / 32) * 128 * 1024);
  float* alpha = (float*)take((size_t)kB * 10 * kT * 4);
  float* beta  = (float*)take((size_t)kB * 10 * kT * 4);
  float* kap   = (float*)take((size_t)kB * 10 * kT * 4);
  u16t* h0 = (u16t*)take((size_t)2 * kB * kH * 2);       // ping-pong bf16 h
  u16t* h1 = (u16t*)take((size_t)2 * kB * kH * 2);
  u16t* h2 = (u16t*)take((size_t)2 * kB * kH * 2);
  float* c0 = (float*)take((size_t)kB * kH * 4);
  float* c1 = (float*)take((size_t)kB * kH * 4);
  float* c2 = (float*)take((size_t)kB * kH * 4);
  (void)c2;

  // zero h/c state (contiguous region h0..c2 end = 786432 bytes)
  const int stateWords = (3 * (2 * kB * kH * 2) + 3 * (kB * kH * 4)) / 4;
  zero_kernel<<<(stateWords + 255) / 256, 256, 0, stream>>>((unsigned*)h0, stateWords);

  // pre-swizzle weights into WMMA B-fragment layout (K = [U ; W ; zero-pad])
  pack_weights_kernel<<<(512 / 32) * 128, 32, 0, stream>>>(U0, 512, nullptr, 0, pk0);
  pack_weights_kernel<<<(1120 / 32) * 128, 32, 0, stream>>>(U1, 512, W1, 595, pk1);
  pack_weights_kernel<<<(1024 / 32) * 128, 32, 0, stream>>>(U2, 512, W2, 512, pk2);

  init_aug_kernel<<<(kB * kT + 255) / 256, 256, 0, stream>>>(stroke, aug);

  // LSTM0: K = 512 (h only), x(3)@W0 folded scalar-wise in epilogue.
  for (int t = 0; t < kT; ++t) {
    const u16t* hr = h0 + (size_t)(t & 1) * kB * kH;
    u16t* hw = h0 + (size_t)((t + 1) & 1) * kB * kH;
    lstm_step_kernel<<<128, 128, 0, stream>>>(
        pk0, 0, hr, hw, c0, nullptr, 0,
        aug + (size_t)t * kB * kAug, kAug, b0, W0, stroke, t);
  }

  abk_kernel<<<(kB * kT * 30 + 255) / 256, 256, 0, stream>>>(aug, Wa, ba, alpha, beta, kap);
  cumsum_kernel<<<(kB * 10 + 63) / 64, 64, 0, stream>>>(kappa0, kap);
  window_kernel<<<kB * kT, 80, 0, stream>>>(alpha, beta, kap, chars, aug);

  // LSTM1: K = 512 (h) + 608 (aug incl. zero pad) = 1120
  for (int t = 0; t < kT; ++t) {
    const u16t* hr = h1 + (size_t)(t & 1) * kB * kH;
    u16t* hw = h1 + (size_t)((t + 1) & 1) * kB * kH;
    lstm_step_kernel<<<128, 128, 0, stream>>>(
        pk1, 608 / 32, hr, hw, c1,
        aug + (size_t)t * kB * kAug, kAug,
        l1o + (size_t)t * kB * kH, kH, b1, nullptr, nullptr, t);
  }

  // LSTM2: K = 512 (h) + 512 (lstm1_out) = 1024
  for (int t = 0; t < kT; ++t) {
    const u16t* hr = h2 + (size_t)(t & 1) * kB * kH;
    u16t* hw = h2 + (size_t)((t + 1) & 1) * kB * kH;
    lstm_step_kernel<<<128, 128, 0, stream>>>(
        pk2, 512 / 32, hr, hw, c2,
        l1o + (size_t)t * kB * kH, kH,
        l2o + (size_t)t * kB * kH, kH, b2, nullptr, nullptr, t);
  }

  mdn_kernel<<<kB * kT, 128, 0, stream>>>(l2o, Wm, bm, out);
}
